// MultiHeadAttention_16724602651061
// MI455X (gfx1250) — matrix operand order repaired and checked by reading, not compiled, not run on hardware
//
#include <hip/hip_runtime.h>

// ---- problem sizes (match reference) ----
#define B_ 4
#define S_ 2048
#define E_ 1024
#define H_ 16
#define D_ 64
#define HD_ (H_*D_)

// attention blocking: 4 waves/block, 2 q-tiles (32 rows) per wave -> 128 q/block
#define WAVES_ 4
#define QW_ 2
#define QB_ (16*QW_*WAVES_)

typedef __attribute__((ext_vector_type(16))) __bf16 v16bf;
typedef __attribute__((ext_vector_type(8)))  __bf16 v8bf;
typedef __attribute__((ext_vector_type(8)))  float  v8f;
typedef __attribute__((ext_vector_type(4)))  unsigned u32x4;
typedef __attribute__((ext_vector_type(8)))  unsigned u32x8;

static __device__ __forceinline__ __bf16 f2bf(float f) {
  union { float f; unsigned u; } a; a.f = f;
  unsigned r = a.u + 0x7FFFu + ((a.u >> 16) & 1u);   // round-to-nearest-even
  unsigned short h = (unsigned short)(r >> 16);
  return __builtin_bit_cast(__bf16, h);
}

static __device__ __forceinline__ v8f zero8() {
  v8f z;
#pragma unroll
  for (int i = 0; i < 8; ++i) z[i] = 0.f;
  return z;
}

// Build a 16-element bf16 fragment from two contiguous 16-byte chunks.
static __device__ __forceinline__ v16bf ldfrag(const __bf16* p0, const __bf16* p1) {
  v8bf lo = *(const v8bf*)p0;
  v8bf hi = *(const v8bf*)p1;
  v16bf r;
#pragma unroll
  for (int i = 0; i < 8; ++i) { r[i] = lo[i]; r[8 + i] = hi[i]; }
  return r;
}

// ---- Tensor Data Mover: DMA a 2-D bf16 tile (tile_d1 rows x tile_d0 elems,
// row stride stride0 elems) from global memory into LDS. Descriptor per
// CDNA5 ISA (D# groups 0/1; groups 2/3 NULL for 2-D).
static __device__ __forceinline__ void tdm_load_2d(
    unsigned lds_addr, const __bf16* gptr,
    unsigned tensor_d0, unsigned tensor_d1,
    unsigned tile_d0, unsigned tile_d1, unsigned stride0)
{
  unsigned long long ga = (unsigned long long)(uintptr_t)gptr;
  u32x4 g0;
  g0[0] = 1u;                                                 // count=1, user D#
  g0[1] = lds_addr;                                           // lds_addr [63:32]
  g0[2] = (unsigned)ga;                                       // global_addr lo
  g0[3] = ((unsigned)(ga >> 32) & 0x01FFFFFFu) | (2u << 30);  // addr hi | type=2
  u32x8 g1;
  g1[0] = 0x00010000u;                                        // data_size=1 (2B)
  g1[1] = (tensor_d0 & 0xFFFFu) << 16;                        // tensor_dim0 lo
  g1[2] = (tensor_d0 >> 16) | ((tensor_d1 & 0xFFFFu) << 16);  // dim0 hi | dim1 lo
  g1[3] = (tensor_d1 >> 16) | (tile_d0 << 16);                // dim1 hi | tile_dim0
  g1[4] = tile_d1 & 0xFFFFu;                                  // tile_dim1 (tile_dim2=0)
  g1[5] = stride0;                                            // tensor_dim0_stride lo
  g1[6] = 0u;                                                 // stride0 hi | stride1 lo
  g1[7] = 0u;                                                 // stride1 hi
  asm volatile("tensor_load_to_lds %0, %1" :: "s"(g0), "s"(g1) : "memory");
}

// ---------------- conversion kernels ----------------
__global__ void cvt_bf16_kernel(const float* __restrict__ in, __bf16* __restrict__ out, int n) {
  int i = blockIdx.x * blockDim.x + threadIdx.x;
  if (i < n) out[i] = f2bf(in[i]);
}

// W [H,E,D] -> WT [H,D,E] (bf16)
__global__ void cvt_wT_kernel(const float* __restrict__ W, __bf16* __restrict__ WT) {
  int i = blockIdx.x * blockDim.x + threadIdx.x;         // over H*E*D
  int d = i % D_; int e = (i / D_) % E_; int h = i / (E_ * D_);
  WT[((size_t)h * D_ + d) * E_ + e] = f2bf(W[i]);
}

// Wo [HD,E] -> WoT [E,HD] (bf16)
__global__ void cvt_woT_kernel(const float* __restrict__ W, __bf16* __restrict__ WT) {
  int i = blockIdx.x * blockDim.x + threadIdx.x;         // over HD*E
  int e = i % E_; int r = i / E_;
  WT[(size_t)e * HD_ + r] = f2bf(W[i]);
}

// ---------------- fused QKV projection ----------------
// one wave per (b, h, 16-row s-tile). Output 16x64 per matrix.
__global__ __launch_bounds__(32) void qkv_kernel(
    const __bf16* __restrict__ xb,                       // [B*S, E]
    const __bf16* __restrict__ wqT, const __bf16* __restrict__ wkT, const __bf16* __restrict__ wvT,
    const float* __restrict__ bq, const float* __restrict__ bk, const float* __restrict__ bv,
    __bf16* __restrict__ Q, __bf16* __restrict__ Kk, __bf16* __restrict__ Vt)
{
  const int lane = threadIdx.x;
  const int half = lane >> 4, l16 = lane & 15;
  const int NST = S_ / 16;
  int tile = blockIdx.x;
  int st = tile % NST; int h = (tile / NST) % H_; int b = tile / (NST * H_);
  int s0 = st * 16;

  const __bf16* xrow = xb + (size_t)(b * S_ + s0 + l16) * E_;
  const __bf16* wT[3] = { wqT + (size_t)h * D_ * E_,
                          wkT + (size_t)h * D_ * E_,
                          wvT + (size_t)h * D_ * E_ };

  v8f acc[3][4];
#pragma unroll
  for (int m = 0; m < 3; ++m)
#pragma unroll
    for (int t = 0; t < 4; ++t) acc[m][t] = zero8();

  for (int e0 = 0; e0 < E_; e0 += 32) {
    // A-frag (16x32 bf16, ISA layout): lanes<16 rows k=[0..7],[16..23]; lanes>=16 k=[8..15],[24..31]
    v16bf a = ldfrag(xrow + e0 + half * 8, xrow + e0 + half * 8 + 16);
#pragma unroll
    for (int m = 0; m < 3; ++m) {
#pragma unroll
      for (int t = 0; t < 4; ++t) {
        // B-frag from transposed weight: lane holds column n=l16, K = half*8..+7, 16+half*8..+7
        const __bf16* wr = wT[m] + (size_t)(t * 16 + l16) * E_ + e0 + half * 8;
        v16bf bw = ldfrag(wr, wr + 16);
        acc[m][t] = __builtin_amdgcn_wmma_f32_16x16x32_bf16(
            false, a, false, bw, (short)0, acc[m][t], false, false);
      }
    }
  }

  size_t baseQ = ((size_t)(b * H_ + h)) * S_ * D_;
  size_t baseV = ((size_t)(b * H_ + h)) * D_ * S_;
#pragma unroll
  for (int t = 0; t < 4; ++t) {
    int d = t * 16 + l16;
    float biq = bq[h * D_ + d], bik = bk[h * D_ + d], biv = bv[h * D_ + d];
#pragma unroll
    for (int v = 0; v < 8; ++v) {
      int s = s0 + v + half * 8;                      // C layout: row = v + 8*half, col = l16
      Q [baseQ + (size_t)s * D_ + d] = f2bf(acc[0][t][v] + biq);
      Kk[baseQ + (size_t)s * D_ + d] = f2bf(acc[1][t][v] + bik);
      Vt[baseV + (size_t)d * S_ + s] = f2bf(acc[2][t][v] + biv);   // store V transposed [D,S]
    }
  }
}

// ---------------- causal flash attention ----------------
// 4 waves per block; each wave owns 2 query tiles (32 rows); 128 queries per
// block share each TDM-loaded K/V chunk (double-buffered in LDS).
__global__ __launch_bounds__(32*WAVES_) void attn_kernel(
    const __bf16* __restrict__ Q, const __bf16* __restrict__ K,
    const __bf16* __restrict__ Vt, __bf16* __restrict__ ctx)
{
  const int tid  = threadIdx.x;
  const int lane = tid & 31;
  const int wid  = tid >> 5;
  const int half = lane >> 4, l16 = lane & 15;
  const int NQB = S_ / QB_;
  int tile = blockIdx.x;
  int qb = tile % NQB; int h = (tile / NQB) % H_; int b = tile / (NQB * H_);
  int bq0 = qb * QB_;
  int q0w = bq0 + wid * (16 * QW_);          // wave covers rows [q0w, q0w+32)

  const __bf16* Qbh = Q  + ((size_t)(b * H_ + h)) * S_ * D_;
  const __bf16* Kbh = K  + ((size_t)(b * H_ + h)) * S_ * D_;
  const __bf16* Vbh = Vt + ((size_t)(b * H_ + h)) * D_ * S_;

  __shared__ __bf16 Kbuf[2][32 * 64];        // 32 keys x 64 d (4 KB each)
  __shared__ __bf16 Vbuf[2][64 * 32];        // 64 d x 32 keys (4 KB each)
  __shared__ __bf16 Pt[WAVES_][QW_][16 * 32];// per-(wave,qtile) P staging

  // resident Q A-frags for both q-tiles, d=0..31 / 32..63
  v16bf aq[QW_][2];
#pragma unroll
  for (int qt = 0; qt < QW_; ++qt) {
    const __bf16* qrow = Qbh + (size_t)(q0w + qt * 16 + l16) * D_;
#pragma unroll
    for (int dc = 0; dc < 2; ++dc)
      aq[qt][dc] = ldfrag(qrow + dc * 32 + half * 8, qrow + dc * 32 + half * 8 + 16);
  }

  float m[QW_][8], l[QW_][8];
  v8f acc[QW_][4];
#pragma unroll
  for (int qt = 0; qt < QW_; ++qt) {
#pragma unroll
    for (int v = 0; v < 8; ++v) { m[qt][v] = -__builtin_inff(); l[qt][v] = 0.f; }
#pragma unroll
    for (int t = 0; t < 4; ++t) acc[qt][t] = zero8();
  }

  const float sc = 0.125f * 1.44269504088896340736f;  // 1/sqrt(D) * log2(e)
  const int nchunk = (bq0 + QB_ + 31) / 32;

  if (wid == 0) {                            // prologue: chunk 0 -> buffer 0
    tdm_load_2d((unsigned)(uintptr_t)&Kbuf[0][0], Kbh, D_, S_, D_, 32, D_);
    tdm_load_2d((unsigned)(uintptr_t)&Vbuf[0][0], Vbh, S_, D_, 32, D_, S_);
  }

  for (int c = 0; c < nchunk; ++c) {
    int kbase = c * 32;
    int cur = c & 1;

    __syncthreads();                         // (A) everyone done reading buf[cur^1]
    if (wid == 0) {
      if (c + 1 < nchunk) {
        int nb = kbase + 32;
        tdm_load_2d((unsigned)(uintptr_t)&Kbuf[cur ^ 1][0], Kbh + (size_t)nb * D_,
                    D_, (unsigned)(S_ - nb), D_, 32, D_);
        tdm_load_2d((unsigned)(uintptr_t)&Vbuf[cur ^ 1][0], Vbh + nb,
                    (unsigned)(S_ - nb), D_, 32, D_, S_);
        __builtin_amdgcn_s_wait_tensorcnt(2);   // chunk c complete
      } else {
        __builtin_amdgcn_s_wait_tensorcnt(0);
      }
    }
    __syncthreads();                         // (B) chunk c visible to all waves
    asm volatile("" ::: "memory");

    bool act[QW_];
#pragma unroll
    for (int qt = 0; qt < QW_; ++qt) act[qt] = (kbase <= q0w + qt * 16 + 15);

    float p[QW_][2][8];
    float bm[QW_][8];
#pragma unroll
    for (int qt = 0; qt < QW_; ++qt)
#pragma unroll
      for (int v = 0; v < 8; ++v) bm[qt][v] = -__builtin_inff();

    // ---- scores: K-frags loaded once per t2, reused by both q-tiles ----
#pragma unroll
    for (int t2 = 0; t2 < 2; ++t2) {
      const __bf16* kr = &Kbuf[cur][(t2 * 16 + l16) * 64];
      v16bf kf0 = ldfrag(kr + half * 8,      kr + half * 8 + 16);
      v16bf kf1 = ldfrag(kr + 32 + half * 8, kr + 32 + half * 8 + 16);
      int key = kbase + t2 * 16 + l16;
#pragma unroll
      for (int qt = 0; qt < QW_; ++qt) {
        if (!act[qt]) continue;
        v8f s = zero8();
        s = __builtin_amdgcn_wmma_f32_16x16x32_bf16(false, aq[qt][0], false, kf0, (short)0, s, false, false);
        s = __builtin_amdgcn_wmma_f32_16x16x32_bf16(false, aq[qt][1], false, kf1, (short)0, s, false, false);
#pragma unroll
        for (int v = 0; v < 8; ++v) {
          int qg = q0w + qt * 16 + v + half * 8;
          float val = (key <= qg) ? s[v] * sc : -__builtin_inff();
          p[qt][t2][v] = val;
          bm[qt][v] = fmaxf(bm[qt][v], val);
        }
      }
    }

    // ---- online softmax + stage P tiles ----
#pragma unroll
    for (int qt = 0; qt < QW_; ++qt) {
      if (!act[qt]) continue;
#pragma unroll
      for (int mask = 1; mask < 16; mask <<= 1)
#pragma unroll
        for (int v = 0; v < 8; ++v) bm[qt][v] = fmaxf(bm[qt][v], __shfl_xor(bm[qt][v], mask, 32));
      float alpha[8], rs[8];
#pragma unroll
      for (int v = 0; v < 8; ++v) {
        float mn = fmaxf(m[qt][v], bm[qt][v]);
        alpha[v] = exp2f(m[qt][v] - mn);
        m[qt][v] = mn;
        float p0 = exp2f(p[qt][0][v] - mn);
        float p1 = exp2f(p[qt][1][v] - mn);
        p[qt][0][v] = p0; p[qt][1][v] = p1;
        rs[v] = p0 + p1;
      }
#pragma unroll
      for (int mask = 1; mask < 16; mask <<= 1)
#pragma unroll
        for (int v = 0; v < 8; ++v) rs[v] += __shfl_xor(rs[v], mask, 32);
#pragma unroll
      for (int v = 0; v < 8; ++v) l[qt][v] = l[qt][v] * alpha[v] + rs[v];
#pragma unroll
      for (int t = 0; t < 4; ++t)
#pragma unroll
        for (int v = 0; v < 8; ++v) acc[qt][t][v] *= alpha[v];
#pragma unroll
      for (int t2 = 0; t2 < 2; ++t2)
#pragma unroll
        for (int v = 0; v < 8; ++v)
          Pt[wid][qt][(v + 8 * half) * 32 + t2 * 16 + l16] = f2bf(p[qt][t2][v]);
    }

    __syncthreads();                         // (C) Pt ordering (also pre-A for next iter)

    // ---- PV: V-frags loaded once, reused by both q-tiles ----
    v16bf vf[4];
#pragma unroll
    for (int t = 0; t < 4; ++t) {
      const __bf16* vp = &Vbuf[cur][(t * 16 + l16) * 32 + half * 8];
      vf[t] = ldfrag(vp, vp + 16);
    }
#pragma unroll
    for (int qt = 0; qt < QW_; ++qt) {
      if (!act[qt]) continue;
      v16bf ap = ldfrag(&Pt[wid][qt][l16 * 32 + half * 8],
                        &Pt[wid][qt][l16 * 32 + half * 8 + 16]);
#pragma unroll
      for (int t = 0; t < 4; ++t)
        acc[qt][t] = __builtin_amdgcn_wmma_f32_16x16x32_bf16(
            false, ap, false, vf[t], (short)0, acc[qt][t], false, false);
    }
  }

  // ---- epilogue: normalize and write concat-head layout ctx [B, S, H*D] ----
  size_t rowbase = (size_t)b * S_ * HD_;
#pragma unroll
  for (int qt = 0; qt < QW_; ++qt) {
#pragma unroll
    for (int t = 0; t < 4; ++t) {
      int d = t * 16 + l16;
#pragma unroll
      for (int v = 0; v < 8; ++v) {
        int s = q0w + qt * 16 + v + half * 8;
        float o = acc[qt][t][v] / l[qt][v];
        ctx[rowbase + (size_t)s * HD_ + h * D_ + d] = f2bf(o);
      }
    }
  }
}

// ---------------- output projection ----------------
// one wave per (16-row tile, 64-col tile) of out = ctx @ Wo + bo
__global__ __launch_bounds__(32) void oproj_kernel(
    const __bf16* __restrict__ ctx, const __bf16* __restrict__ WoT,
    const float* __restrict__ bo, float* __restrict__ out)
{
  const int lane = threadIdx.x;
  const int half = lane >> 4, l16 = lane & 15;
  const int NE = E_ / 64;
  int tile = blockIdx.x;
  int nt = tile % NE; int rt = tile / NE;
  int r0 = rt * 16, n0 = nt * 64;

  const __bf16* arow = ctx + (size_t)(r0 + l16) * HD_;
  v8f acc[4];
#pragma unroll
  for (int t = 0; t < 4; ++t) acc[t] = zero8();

  for (int k0 = 0; k0 < HD_; k0 += 32) {
    v16bf a = ldfrag(arow + k0 + half * 8, arow + k0 + half * 8 + 16);
#pragma unroll
    for (int t = 0; t < 4; ++t) {
      const __bf16* wr = WoT + (size_t)(n0 + t * 16 + l16) * HD_ + k0 + half * 8;
      v16bf bw = ldfrag(wr, wr + 16);
      acc[t] = __builtin_amdgcn_wmma_f32_16x16x32_bf16(
          false, a, false, bw, (short)0, acc[t], false, false);
    }
  }
#pragma unroll
  for (int t = 0; t < 4; ++t) {
    int col = n0 + t * 16 + l16;
    float bb = bo[col];
#pragma unroll
    for (int v = 0; v < 8; ++v) {
      int r = r0 + v + 8 * half;
      out[(size_t)r * E_ + col] = acc[t][v] + bb;
    }
  }
}

// ---------------- launcher ----------------
extern "C" void kernel_launch(void* const* d_in, const int* in_sizes, int n_in,
                              void* d_out, int out_size, void* d_ws, size_t ws_size,
                              hipStream_t stream) {
  const float* x  = (const float*)d_in[0];
  const float* Wq = (const float*)d_in[1];
  const float* Wk = (const float*)d_in[2];
  const float* Wv = (const float*)d_in[3];
  const float* bq = (const float*)d_in[4];
  const float* bk = (const float*)d_in[5];
  const float* bv = (const float*)d_in[6];
  const float* Wo = (const float*)d_in[7];
  const float* bo = (const float*)d_in[8];
  float* out = (float*)d_out;
  (void)in_sizes; (void)n_in; (void)out_size; (void)ws_size;

  char* ws = (char*)d_ws;
  size_t off = 0;
  auto alloc = [&](size_t bytes) -> char* {
    char* p = ws + off;
    off += (bytes + 255) & ~(size_t)255;
    return p;
  };
  __bf16* xb   = (__bf16*)alloc((size_t)B_ * S_ * E_ * 2);
  __bf16* wqT  = (__bf16*)alloc((size_t)H_ * D_ * E_ * 2);
  __bf16* wkT  = (__bf16*)alloc((size_t)H_ * D_ * E_ * 2);
  __bf16* wvT  = (__bf16*)alloc((size_t)H_ * D_ * E_ * 2);
  __bf16* woT  = (__bf16*)alloc((size_t)E_ * HD_ * 2);
  __bf16* Qb   = (__bf16*)alloc((size_t)B_ * H_ * S_ * D_ * 2);
  __bf16* Kb   = (__bf16*)alloc((size_t)B_ * H_ * S_ * D_ * 2);
  __bf16* Vtb  = (__bf16*)alloc((size_t)B_ * H_ * D_ * S_ * 2);
  __bf16* ctxb = (__bf16*)alloc((size_t)B_ * S_ * HD_ * 2);

  int n = B_ * S_ * E_;
  cvt_bf16_kernel<<<(n + 255) / 256, 256, 0, stream>>>(x, xb, n);
  int nw = H_ * E_ * D_;
  cvt_wT_kernel<<<nw / 256, 256, 0, stream>>>(Wq, wqT);
  cvt_wT_kernel<<<nw / 256, 256, 0, stream>>>(Wk, wkT);
  cvt_wT_kernel<<<nw / 256, 256, 0, stream>>>(Wv, wvT);
  cvt_woT_kernel<<<(HD_ * E_) / 256, 256, 0, stream>>>(Wo, woT);

  qkv_kernel<<<B_ * H_ * (S_ / 16), 32, 0, stream>>>(
      xb, wqT, wkT, wvT, bq, bk, bv, Qb, Kb, Vtb);

  attn_kernel<<<B_ * H_ * (S_ / QB_), 32 * WAVES_, 0, stream>>>(Qb, Kb, Vtb, ctxb);

  oproj_kernel<<<(B_ * S_ / 16) * (E_ / 64), 32, 0, stream>>>(ctxb, woT, bo, out);
}
